// GRU_41712722378994
// MI455X (gfx1250) — matrix operand order repaired and checked by reading, not compiled, not run on hardware
//
#include <hip/hip_runtime.h>

// ---------- types matching gfx1250 WMMA operand layouts ----------
typedef __attribute__((ext_vector_type(16))) _Float16 v16h;
typedef __attribute__((ext_vector_type(8)))  _Float16 v8h;
typedef __attribute__((ext_vector_type(8)))  float    v8f;

struct V8x2 { v8h lo; v8h hi; };

static __device__ __forceinline__ v8f zero8() {
  v8f z;
#pragma unroll
  for (int i = 0; i < 8; ++i) z[i] = 0.f;
  return z;
}

// ---------------------------------------------------------------------------
// One 16x16 output tile: C[16,16] = A(16xK) * B(16xK)^T, A/B row-major f16,
// C fp32 (overwritten). K must be a multiple of 32. Uses v_wmma_f32_16x16x32_f16.
// A frag: lanes 0-15 hold K 0..15 of row M=lane; lanes 16-31
// hold K 16..31.  B frag: lanes 0-15 hold K 0..15 of col N=lane,
// lanes 16-31 hold K 16..31.
// ---------------------------------------------------------------------------
static __device__ __forceinline__ void wmma_tile16(
    const _Float16* __restrict__ A, int lda,
    const _Float16* __restrict__ B, int ldb,
    float* __restrict__ C, int ldc, int K, int lane)
{
  const int half = lane >> 4;
  const int l16  = lane & 15;
  const _Float16* ap = A + (size_t)l16 * lda + half * 16;
  const _Float16* bp = B + (size_t)l16 * ldb + half * 16;
  v8f acc = zero8();
  for (int k = 0; k < K; k += 32) {
    v8h lo = *(const v8h*)(ap + k);
    v8h hi = *(const v8h*)(ap + k + 8);
    v16h a = __builtin_bit_cast(v16h, V8x2{lo, hi});
    v16h b = *(const v16h*)(bp + k);
    acc = __builtin_amdgcn_wmma_f32_16x16x32_f16(false, a, false, b,
                                                 (short)0, acc, false, false);
  }
  float* cp = C + (size_t)(half * 8) * ldc + l16;
#pragma unroll
  for (int v = 0; v < 8; ++v) cp[(size_t)v * ldc] = acc[v];
}

// ---------------------------------------------------------------------------
// Big GEMM: C[M,N] = A(MxK,f16) * B(NxK,f16)^T (+bias).  Block = 128x128 tile,
// 8 waves as 2(M)x4(N); each wave: 4 M-tiles x 2 N-tiles (8 accumulators).
// M,N multiples of 128; K multiple of 32 (true for all uses here).
// ---------------------------------------------------------------------------
__global__ void __launch_bounds__(256) gemm_f16_nt(
    const _Float16* __restrict__ A, const _Float16* __restrict__ B,
    float* __restrict__ C, const float* __restrict__ bias,
    int M, int N, int K, int nbn)
{
  const int bm   = blockIdx.x / nbn;
  const int bn   = blockIdx.x % nbn;
  const int wid  = threadIdx.x >> 5;
  const int lane = threadIdx.x & 31;
  const int wm = wid >> 2, wn = wid & 3;
  const int half = lane >> 4, l16 = lane & 15;
  const int mBase = bm * 128 + wm * 64;
  const int nBase = bn * 128 + wn * 32;

  const _Float16* ap[4];
  const _Float16* bp[2];
#pragma unroll
  for (int mi = 0; mi < 4; ++mi)
    ap[mi] = A + (size_t)(mBase + mi * 16 + l16) * K + half * 16;
#pragma unroll
  for (int ni = 0; ni < 2; ++ni)
    bp[ni] = B + (size_t)(nBase + ni * 16 + l16) * K + half * 16;

  v8f acc[4][2];
#pragma unroll
  for (int mi = 0; mi < 4; ++mi)
#pragma unroll
    for (int ni = 0; ni < 2; ++ni) acc[mi][ni] = zero8();

  for (int k = 0; k < K; k += 32) {
    v16h a[4], b[2];
#pragma unroll
    for (int mi = 0; mi < 4; ++mi) {
      v8h lo = *(const v8h*)(ap[mi] + k);
      v8h hi = *(const v8h*)(ap[mi] + k + 8);
      a[mi] = __builtin_bit_cast(v16h, V8x2{lo, hi});
    }
#pragma unroll
    for (int ni = 0; ni < 2; ++ni) b[ni] = *(const v16h*)(bp[ni] + k);
#pragma unroll
    for (int mi = 0; mi < 4; ++mi)
#pragma unroll
      for (int ni = 0; ni < 2; ++ni)
        acc[mi][ni] = __builtin_amdgcn_wmma_f32_16x16x32_f16(
            false, a[mi], false, b[ni], (short)0, acc[mi][ni], false, false);
  }

#pragma unroll
  for (int mi = 0; mi < 4; ++mi)
#pragma unroll
    for (int ni = 0; ni < 2; ++ni) {
      const int n = nBase + ni * 16 + l16;
      const float bb = bias ? bias[n] : 0.f;
      float* cp = C + (size_t)(mBase + mi * 16 + half * 8) * N + n;
#pragma unroll
      for (int v = 0; v < 8; ++v) cp[(size_t)v * N] = acc[mi][ni][v] + bb;
    }
}

// ---------------------------------------------------------------------------
// Device-wide sense barrier (bar[0]=count, bar[1]=generation).
// ---------------------------------------------------------------------------
static __device__ __forceinline__ void grid_barrier(unsigned* bar, unsigned nblk)
{
  __syncthreads();
  if (threadIdx.x == 0) {
    __threadfence();
    volatile unsigned* gen = bar + 1;
    const unsigned g = *gen;
    if (atomicAdd(bar, 1u) == nblk - 1u) {
      *((volatile unsigned*)bar) = 0u;
      __threadfence();
      atomicAdd(bar + 1, 1u);
    } else {
      while (*gen == g) { __builtin_amdgcn_s_sleep(2); }
    }
    __threadfence();
  }
  __syncthreads();
}

// ---------------------------------------------------------------------------
// LN + sigmoid + candidate LN + tanh + gate update for one batch row (1 wave).
// xrow: [0..1023]=x-side rz pre-acts, [1024..1535]=x-side h pre-acts.
// hrow: [0..1023]=h_prev@u_rz^T, [1024..1535]=h_prev@u_h^T.
// ---------------------------------------------------------------------------
static __device__ __forceinline__ void gru_gates(
    int lane, const float* __restrict__ xrow, const float* __restrict__ hrow,
    const float* __restrict__ b_rz, const float* __restrict__ b_h,
    const float* __restrict__ g_rz, const float* __restrict__ be_rz,
    const float* __restrict__ g_h,  const float* __restrict__ be_h,
    float* __restrict__ hf, _Float16* __restrict__ h16,
    _Float16* __restrict__ yrow, float* __restrict__ hid)
{
  float rv[32], sg[32];
  float s1 = 0.f, s2 = 0.f;
#pragma unroll
  for (int i = 0; i < 32; ++i) {
    const int j = i * 32 + lane;
    const float v = xrow[j] + hrow[j] + b_rz[j];
    rv[i] = v; s1 += v; s2 += v * v;
  }
#pragma unroll
  for (int off = 16; off > 0; off >>= 1) {
    s1 += __shfl_xor(s1, off, 32);
    s2 += __shfl_xor(s2, off, 32);
  }
  float mu  = s1 * (1.f / 1024.f);
  float inv = rsqrtf(s2 * (1.f / 1024.f) - mu * mu + 1e-5f);
#pragma unroll
  for (int i = 0; i < 32; ++i) {
    const int j = i * 32 + lane;
    const float v = (rv[i] - mu) * inv * g_rz[j] + be_rz[j];
    sg[i] = 1.f / (1.f + __expf(-v));          // i<16 -> r_k, i>=16 -> z_k
  }
  float hv[16];
  s1 = 0.f; s2 = 0.f;
#pragma unroll
  for (int i = 0; i < 16; ++i) {
    const int k = i * 32 + lane;
    const float v = xrow[1024 + k] + b_h[k] + sg[i] * hrow[1024 + k];
    hv[i] = v; s1 += v; s2 += v * v;
  }
#pragma unroll
  for (int off = 16; off > 0; off >>= 1) {
    s1 += __shfl_xor(s1, off, 32);
    s2 += __shfl_xor(s2, off, 32);
  }
  mu  = s1 * (1.f / 512.f);
  inv = rsqrtf(s2 * (1.f / 512.f) - mu * mu + 1e-5f);
#pragma unroll
  for (int i = 0; i < 16; ++i) {
    const int k = i * 32 + lane;
    const float ht = tanhf((hv[i] - mu) * inv * g_h[k] + be_h[k]);
    const float z  = sg[i + 16];
    const float hn = (1.f - z) * hf[k] + z * ht;
    hf[k]  = hn;
    h16[k] = (_Float16)hn;
    if (yrow) yrow[k] = (_Float16)hn;
    if (hid)  hid[k]  = hn;
  }
}

// ---------------------------------------------------------------------------
// Persistent recurrent kernel: 96 blocks x 256 threads = 768 waves.
// Per step: P1 = [h0@U0cat^T | h1@U1cat^T] (768 tiles, 1/wave), gates L0,
// G1 = h0_new@W1cat^T (384 tiles), gates L1.  4 grid barriers / step.
// ---------------------------------------------------------------------------
__global__ void __launch_bounds__(256) gru_persistent(
    const float* __restrict__ Xproj, float* __restrict__ PH1, float* __restrict__ G1,
    float* h0f, float* h1f, _Float16* h0h, _Float16* h1h,
    const _Float16* __restrict__ U0, const _Float16* __restrict__ U1,
    const _Float16* __restrict__ W1,
    _Float16* __restrict__ Yseq, float* __restrict__ hid_out,
    const float* b_rz0, const float* b_h0, const float* g_rz0, const float* be_rz0,
    const float* g_h0,  const float* be_h0,
    const float* b_rz1, const float* b_h1, const float* g_rz1, const float* be_rz1,
    const float* g_h1,  const float* be_h1,
    unsigned* bar)
{
  const int wid   = threadIdx.x >> 5;
  const int lane  = threadIdx.x & 31;
  const int gwave = blockIdx.x * 8 + wid;        // 0..767
  const int tm = gwave & 3;                      // M tile (batch 64 = 4 tiles)
  const int tn = gwave >> 2;                     // N tile

  for (int t = 0; t < 512; ++t) {
    // ---- Phase 1: recurrent GEMMs for both layers (h from previous step) ----
    {
      const int m0 = tm * 16, n0 = tn * 16;      // n0 in [0,3072)
      const _Float16* A;
      const _Float16* B;
      if (n0 < 1536) { A = h0h + m0 * 512; B = U0 + (size_t)n0 * 512; }
      else           { A = h1h + m0 * 512; B = U1 + (size_t)(n0 - 1536) * 512; }
      wmma_tile16(A, 512, B, 512, PH1 + (size_t)m0 * 3072 + n0, 3072, 512, lane);
    }
    grid_barrier(bar, 96);

    // ---- Phase 2: layer-0 LN/gates (one wave per batch row) ----
    if (gwave < 64) {
      const int b = gwave;
      gru_gates(lane,
                Xproj + (size_t)(b * 512 + t) * 1536,
                PH1 + (size_t)b * 3072,
                b_rz0, b_h0, g_rz0, be_rz0, g_h0, be_h0,
                h0f + b * 512, h0h + b * 512,
                (_Float16*)nullptr,
                (t == 511) ? hid_out + b * 512 : nullptr);
    }
    grid_barrier(bar, 96);

    // ---- Phase 3: layer-1 input GEMM  G1 = h0_new @ [w_rz1;w_h1]^T ----
    if (gwave < 384) {
      const int m0 = tm * 16, n0 = tn * 16;      // n0 in [0,1536)
      wmma_tile16(h0h + m0 * 512, 512, W1 + (size_t)n0 * 512, 512,
                  G1 + (size_t)m0 * 1536 + n0, 1536, 512, lane);
    }
    grid_barrier(bar, 96);

    // ---- Phase 4: layer-1 LN/gates, emit y_t ----
    if (gwave < 64) {
      const int b = gwave;
      gru_gates(lane,
                G1 + (size_t)b * 1536,
                PH1 + (size_t)b * 3072 + 1536,
                b_rz1, b_h1, g_rz1, be_rz1, g_h1, be_h1,
                h1f + b * 512, h1h + b * 512,
                Yseq + (size_t)(b * 512 + t) * 512,
                (t == 511) ? hid_out + 32768 + b * 512 : nullptr);
    }
    grid_barrier(bar, 96);
  }
}

// ---------- small utility kernels ----------
__global__ void zero_u32(unsigned* p, int n) {
  for (int i = blockIdx.x * blockDim.x + threadIdx.x; i < n;
       i += gridDim.x * blockDim.x)
    p[i] = 0u;
}

__global__ void f32_to_f16(const float* __restrict__ s, _Float16* __restrict__ d, int n) {
  for (int i = blockIdx.x * blockDim.x + threadIdx.x; i < n;
       i += gridDim.x * blockDim.x)
    d[i] = (_Float16)s[i];
}

// ---------------------------------------------------------------------------
extern "C" void kernel_launch(void* const* d_in, const int* in_sizes, int n_in,
                              void* d_out, int out_size, void* d_ws, size_t ws_size,
                              hipStream_t stream)
{
  (void)in_sizes; (void)n_in; (void)out_size; (void)ws_size;
  const float* x      = (const float*)d_in[0];
  const float* fc_w   = (const float*)d_in[1];
  const float* fc_b   = (const float*)d_in[2];
  const float* w_rz0  = (const float*)d_in[3];
  const float* b_rz0  = (const float*)d_in[4];
  const float* u_rz0  = (const float*)d_in[5];
  const float* w_h0   = (const float*)d_in[6];
  const float* b_h0   = (const float*)d_in[7];
  const float* u_h0   = (const float*)d_in[8];
  const float* gr0    = (const float*)d_in[9];
  const float* br0    = (const float*)d_in[10];
  const float* gh0    = (const float*)d_in[11];
  const float* bh0    = (const float*)d_in[12];
  const float* w_rz1  = (const float*)d_in[13];
  const float* b_rz1  = (const float*)d_in[14];
  const float* u_rz1  = (const float*)d_in[15];
  const float* w_h1   = (const float*)d_in[16];
  const float* b_h1   = (const float*)d_in[17];
  const float* u_h1   = (const float*)d_in[18];
  const float* gr1    = (const float*)d_in[19];
  const float* br1    = (const float*)d_in[20];
  const float* gh1    = (const float*)d_in[21];
  const float* bh1    = (const float*)d_in[22];
  float* out = (float*)d_out;

  char* W = (char*)d_ws;
  size_t off = 0;
  auto nxt = [&](size_t bytes) {
    size_t o = off;
    off += (bytes + 255) & ~(size_t)255;
    return o;
  };
  const size_t XH   = nxt((size_t)16777216 * 2);      // x as f16 (32768 x 512)
  const size_t W0   = nxt((size_t)1536 * 512 * 2);    // [w_rz0; w_h0]
  const size_t U0   = nxt((size_t)1536 * 512 * 2);    // [u_rz0; u_h0]
  const size_t W1   = nxt((size_t)1536 * 512 * 2);    // [w_rz1; w_h1]
  const size_t U1   = nxt((size_t)1536 * 512 * 2);    // [u_rz1; u_h1]
  const size_t FCW  = nxt((size_t)512 * 512 * 2);     // fc_w f16
  const size_t XPRJ = nxt((size_t)32768 * 1536 * 4);  // x-projections, fp32
  const size_t PH1  = nxt((size_t)64 * 3072 * 4);     // per-step recurrent accs
  const size_t G1O  = nxt((size_t)64 * 1536 * 4);     // per-step layer-1 x-accs
  const size_t H0F  = nxt((size_t)64 * 512 * 4);
  const size_t H1F  = nxt((size_t)64 * 512 * 4);
  const size_t H0H  = nxt((size_t)64 * 512 * 2);
  const size_t H1H  = nxt((size_t)64 * 512 * 2);
  const size_t BAR  = nxt(256);
  const size_t YSEQ = nxt((size_t)32768 * 512 * 2);   // layer-1 outputs, f16

  // 1) zero h-states + barrier (contiguous: H0F..BAR, all 256-aligned sizes)
  {
    const int words = (int)((131072 + 131072 + 65536 + 65536 + 256) / 4);
    zero_u32<<<(words + 255) / 256, 256, 0, stream>>>((unsigned*)(W + H0F), words);
  }

  // 2) f32 -> f16 conversions
  auto conv = [&](const float* src, _Float16* dst, int n) {
    int blk = (n + 255) / 256;
    if (blk > 16384) blk = 16384;
    f32_to_f16<<<blk, 256, 0, stream>>>(src, dst, n);
  };
  conv(x,     (_Float16*)(W + XH),               16777216);
  conv(w_rz0, (_Float16*)(W + W0),               524288);
  conv(w_h0,  (_Float16*)(W + W0) + 524288,      262144);
  conv(u_rz0, (_Float16*)(W + U0),               524288);
  conv(u_h0,  (_Float16*)(W + U0) + 524288,      262144);
  conv(w_rz1, (_Float16*)(W + W1),               524288);
  conv(w_h1,  (_Float16*)(W + W1) + 524288,      262144);
  conv(u_rz1, (_Float16*)(W + U1),               524288);
  conv(u_h1,  (_Float16*)(W + U1) + 524288,      262144);
  conv(fc_w,  (_Float16*)(W + FCW),              262144);

  // 3) hoisted input-projection GEMM: Xproj = x_f16 @ [w_rz0;w_h0]^T
  gemm_f16_nt<<<(32768 / 128) * (1536 / 128), 256, 0, stream>>>(
      (const _Float16*)(W + XH), (const _Float16*)(W + W0),
      (float*)(W + XPRJ), nullptr, 32768, 1536, 512, 1536 / 128);

  // 4) persistent recurrence
  gru_persistent<<<96, 256, 0, stream>>>(
      (const float*)(W + XPRJ), (float*)(W + PH1), (float*)(W + G1O),
      (float*)(W + H0F), (float*)(W + H1F),
      (_Float16*)(W + H0H), (_Float16*)(W + H1H),
      (const _Float16*)(W + U0), (const _Float16*)(W + U1),
      (const _Float16*)(W + W1),
      (_Float16*)(W + YSEQ), out + 16777216,
      b_rz0, b_h0, gr0, br0, gh0, bh0,
      b_rz1, b_h1, gr1, br1, gh1, bh1,
      (unsigned*)(W + BAR));

  // 5) final FC: logits = Yseq @ fc_w^T + fc_b  (straight into d_out)
  gemm_f16_nt<<<(32768 / 128) * (512 / 128), 256, 0, stream>>>(
      (const _Float16*)(W + YSEQ), (const _Float16*)(W + FCW),
      out, fc_b, 32768, 512, 512, 512 / 128);
}
